// BahdanauAttention_40973988004681
// MI455X (gfx1250) — matrix operand order repaired and checked by reading, not compiled, not run on hardware
//
#include <hip/hip_runtime.h>
#include <hip/hip_bf16.h>

// ---------------------------------------------------------------------------
// Bahdanau attention, MI455X (gfx1250), wave32 + WMMA bf16.
// T=2048, B=32, H=K=Q=1024.
//   d_in[0] decoder_hidden f32 [1,32,1024]
//   d_in[1] encoder_hidden f32 [2048,32,1024]
//   d_in[2] mask          i32 [2048,32]
//   d_in[3] Wk            f32 [1024,1024]
//   d_in[4] Wq            f32 [1024,1024]
//   d_in[5] v             f32 [1024]
//   d_out: context f32 [32*1024] ++ alphas f32 [2048*32]
// Workspace: [0,2MB) Wk bf16 ; [2MB, +128KB) query f32 ; then scores f32 256KB
// ---------------------------------------------------------------------------

typedef __attribute__((ext_vector_type(16))) __bf16 v16bf;
typedef __attribute__((ext_vector_type(8)))  float  v8f;

#define TT 2048
#define BB 32
#define HH 1024
#define KK 1024

// Native CDNA5 transcendental tanh (V_TANH_F32) when the builtin exists;
// fall back to libm expansion otherwise.
#if defined(__has_builtin)
#  if __has_builtin(__builtin_amdgcn_tanhf)
#    define FAST_TANH(x) __builtin_amdgcn_tanhf(x)
#  elif __has_builtin(__builtin_amdgcn_tanh_f32)
#    define FAST_TANH(x) __builtin_amdgcn_tanh_f32(x)
#  else
#    define FAST_TANH(x) tanhf(x)
#  endif
#else
#  define FAST_TANH(x) tanhf(x)
#endif

__device__ __forceinline__ unsigned int pack_bf16(float a, float b) {
  unsigned ua = __builtin_bit_cast(unsigned, a);
  unsigned ub = __builtin_bit_cast(unsigned, b);
  ua = (ua + 0x7FFFu + ((ua >> 16) & 1u)) >> 16;   // round-to-nearest-even
  ub = (ub + 0x7FFFu + ((ub >> 16) & 1u)) >> 16;
  return ua | (ub << 16);
}

union V16BF { v16bf v; uint4 u[2]; };

__device__ __forceinline__ v16bf ld_bf16x16(const void* p) {
  V16BF t;
  const uint4* q = (const uint4*)p;
  t.u[0] = q[0];
  t.u[1] = q[1];
  return t.v;
}

// ---------------- Kernel A: Wk f32 -> bf16 (same [H][K] layout) -------------
__global__ __launch_bounds__(256) void cvt_wk_kernel(const float* __restrict__ Wk,
                                                     unsigned int* __restrict__ out) {
  int i = blockIdx.x * 256 + threadIdx.x;       // pair index, 0..524287
  float2 f = ((const float2*)Wk)[i];
  out[i] = pack_bf16(f.x, f.y);
}

// ---------------- Kernel B: query[b][h] = dec[b] . Wq[h] --------------------
__global__ __launch_bounds__(256) void query_kernel(const float* __restrict__ dec,
                                                    const float* __restrict__ Wq,
                                                    float* __restrict__ query) {
  __shared__ float sdec[32 * 256];
  int h = blockIdx.x * 256 + threadIdx.x;       // 0..1023
  const float* wrow = Wq + (size_t)h * 1024;
  float acc[32];
#pragma unroll
  for (int b = 0; b < 32; ++b) acc[b] = 0.0f;

  for (int qc = 0; qc < 4; ++qc) {
    __syncthreads();
    for (int i = threadIdx.x; i < 32 * 256; i += 256) {
      int b = i >> 8, q = i & 255;
      sdec[b * 256 + q] = dec[b * 1024 + qc * 256 + q];
    }
    __syncthreads();
    for (int q = 0; q < 256; ++q) {
      float w = wrow[qc * 256 + q];
#pragma unroll
      for (int b = 0; b < 32; ++b) acc[b] = fmaf(w, sdec[b * 256 + q], acc[b]);
    }
  }
#pragma unroll
  for (int b = 0; b < 32; ++b) query[b * 1024 + h] = acc[b];
}

// ---------------- Kernel C: fused GEMM + tanh + v-dot -> scores -------------
// Block: 128 threads (4 waves). Each wave handles 32 rows m = t*B + b.
// Dynamic LDS: 256 KB (4 waves x 64 KB of bf16-swizzled A).
__global__ __launch_bounds__(128) void scores_kernel(
    const float* __restrict__ enc,            // [65536][1024] f32
    const unsigned short* __restrict__ wkbf,  // [1024][1024] bf16
    const float* __restrict__ query,          // [32][1024] f32
    const float* __restrict__ vvec,           // [1024] f32
    float* __restrict__ scores)               // [65536] f32
{
  extern __shared__ unsigned int ldsA[];      // 65536 uints = 256 KB
  const int tid  = threadIdx.x;
  const int wave = tid >> 5;
  const int lane = tid & 31;
  const int g    = lane >> 4;                 // lane half: 0 or 1
  const int L    = lane & 15;
  const int m_base = blockIdx.x * 128 + wave * 32;   // multiple of 32

  unsigned int* myLds = ldsA + wave * 16384;  // 64 KB per wave

  // ---- Stage A (2 tiles of 16 rows x 1024 k) as bf16 in WMMA-A swizzle ----
  // Per tile tau, chunk kc, lane (L+16g) stores 16 halves:
  //   K = 32*kc + 8g + {0..7}  then  K = 32*kc + 16 + 8g + {0..7}
#pragma unroll 1
  for (int tau = 0; tau < 2; ++tau) {
    const float* rowp = enc + (size_t)(m_base + tau * 16 + L) * 1024;
#pragma unroll 4
    for (int kc = 0; kc < 32; ++kc) {
      int k1 = 32 * kc + 8 * g;
      int k2 = k1 + 16;
      float4 f0 = *(const float4*)(rowp + k1);
      float4 f1 = *(const float4*)(rowp + k1 + 4);
      float4 f2 = *(const float4*)(rowp + k2);
      float4 f3 = *(const float4*)(rowp + k2 + 4);
      unsigned int* dst = myLds + tau * 8192 + kc * 256 + lane * 8;
      dst[0] = pack_bf16(f0.x, f0.y);
      dst[1] = pack_bf16(f0.z, f0.w);
      dst[2] = pack_bf16(f1.x, f1.y);
      dst[3] = pack_bf16(f1.z, f1.w);
      dst[4] = pack_bf16(f2.x, f2.y);
      dst[5] = pack_bf16(f2.z, f2.w);
      dst[6] = pack_bf16(f3.x, f3.y);
      dst[7] = pack_bf16(f3.z, f3.w);
    }
  }
  __syncthreads();

  float sAcc0[8], sAcc1[8];
#pragma unroll
  for (int i = 0; i < 8; ++i) { sAcc0[i] = 0.0f; sAcc1[i] = 0.0f; }

  // ---- Sweep the 64 n-tiles of H; K-loop of 32 WMMAs per tile -------------
#pragma unroll 1
  for (int nt = 0; nt < 64; ++nt) {
    const int n = nt * 16 + L;
    const unsigned short* bcol = wkbf + (size_t)n * 1024 + 8 * g;
    const float vn = vvec[n];

    v8f acc0 = {0.f, 0.f, 0.f, 0.f, 0.f, 0.f, 0.f, 0.f};
    v8f acc1 = {0.f, 0.f, 0.f, 0.f, 0.f, 0.f, 0.f, 0.f};

#pragma unroll 4
    for (int kc = 0; kc < 32; ++kc) {
      V16BF bt; bt.u[0] = *(const uint4*)(bcol + 32 * kc); bt.u[1] = *(const uint4*)(bcol + 32 * kc + 16); v16bf bm = bt.v; // B 32x16 tile
      v16bf a0 = ld_bf16x16(myLds +        kc * 256 + lane * 8); // A tile 0
      v16bf a1 = ld_bf16x16(myLds + 8192 + kc * 256 + lane * 8); // A tile 1
      acc0 = __builtin_amdgcn_wmma_f32_16x16x32_bf16(false, a0, false, bm,
                                                     (short)0, acc0, false, false);
      acc1 = __builtin_amdgcn_wmma_f32_16x16x32_bf16(false, a1, false, bm,
                                                     (short)0, acc1, false, false);
    }

    // Epilogue: C VGPR i holds rows r=i+8g (lanes split at 16), col n.
    // Single V_TANH_F32 per element keeps WMMA as the dominant issue consumer.
#pragma unroll
    for (int i = 0; i < 8; ++i) {
      const int r = i + 8 * g;
      float q0 = query[(r)      * 1024 + n];   // b = tau*16 + r, tau=0
      float q1 = query[(16 + r) * 1024 + n];   // tau=1
      sAcc0[i] = fmaf(vn, FAST_TANH(acc0[i] + q0), sAcc0[i]);
      sAcc1[i] = fmaf(vn, FAST_TANH(acc1[i] + q1), sAcc1[i]);
    }
  }

  // ---- Reduce over the 16 lanes of each half, write scores ----------------
#pragma unroll
  for (int i = 0; i < 8; ++i) {
    float s0 = sAcc0[i], s1 = sAcc1[i];
#pragma unroll
    for (int m = 1; m <= 8; m <<= 1) {
      s0 += __shfl_xor(s0, m, 32);
      s1 += __shfl_xor(s1, m, 32);
    }
    if (L == 0) {
      const int r = i + 8 * g;
      scores[m_base + r]      = s0;
      scores[m_base + 16 + r] = s1;
    }
  }
}

// ---------------- Kernel D: masked softmax over T per column b --------------
__global__ __launch_bounds__(256) void softmax_kernel(const float* __restrict__ scores,
                                                      const int* __restrict__ mask,
                                                      float* __restrict__ alphas) {
  __shared__ float red[256];
  const int b = blockIdx.x;
  const int tid = threadIdx.x;
  float vals[8];
  float mx = -INFINITY;
#pragma unroll
  for (int i = 0; i < 8; ++i) {
    int t = i * 256 + tid;
    float s = scores[t * 32 + b];
    s = (mask[t * 32 + b] == 0) ? -INFINITY : s;
    vals[i] = s;
    mx = fmaxf(mx, s);
  }
  red[tid] = mx; __syncthreads();
  for (int off = 128; off > 0; off >>= 1) {
    if (tid < off) red[tid] = fmaxf(red[tid], red[tid + off]);
    __syncthreads();
  }
  mx = red[0]; __syncthreads();

  float sum = 0.0f;
#pragma unroll
  for (int i = 0; i < 8; ++i) {
    float e = __expf(vals[i] - mx);
    vals[i] = e;
    sum += e;
  }
  red[tid] = sum; __syncthreads();
  for (int off = 128; off > 0; off >>= 1) {
    if (tid < off) red[tid] += red[tid + off];
    __syncthreads();
  }
  const float inv = 1.0f / red[0];
#pragma unroll
  for (int i = 0; i < 8; ++i)
    alphas[(i * 256 + tid) * 32 + b] = vals[i] * inv;
}

// ---------------- Kernel E: context[b][k] = sum_t alpha[t][b]*enc[t][b][k] --
__global__ __launch_bounds__(256) void context_kernel(const float* __restrict__ enc,
                                                      const float* __restrict__ alphas,
                                                      float* __restrict__ ctx) {
  __shared__ float al[TT];
  const int k = blockIdx.x * 256 + threadIdx.x;
  const int b = blockIdx.y;
  for (int t = threadIdx.x; t < TT; t += 256) al[t] = alphas[t * 32 + b];
  __syncthreads();

  const float* base = enc + (size_t)b * 1024 + k;
  float a0 = 0.f, a1 = 0.f, a2 = 0.f, a3 = 0.f;
#pragma unroll 1
  for (int t = 0; t < TT; t += 4) {
    a0 = fmaf(al[t + 0], base[(size_t)(t + 0) * 32768], a0);
    a1 = fmaf(al[t + 1], base[(size_t)(t + 1) * 32768], a1);
    a2 = fmaf(al[t + 2], base[(size_t)(t + 2) * 32768], a2);
    a3 = fmaf(al[t + 3], base[(size_t)(t + 3) * 32768], a3);
  }
  ctx[b * 1024 + k] = (a0 + a1) + (a2 + a3);
}

// ---------------------------------------------------------------------------
extern "C" void kernel_launch(void* const* d_in, const int* in_sizes, int n_in,
                              void* d_out, int out_size, void* d_ws, size_t ws_size,
                              hipStream_t stream) {
  const float* dec  = (const float*)d_in[0];
  const float* enc  = (const float*)d_in[1];
  const int*   mask = (const int*)d_in[2];
  const float* Wk   = (const float*)d_in[3];
  const float* Wq   = (const float*)d_in[4];
  const float* vv   = (const float*)d_in[5];

  char* ws = (char*)d_ws;
  unsigned int*   wk_bf16 = (unsigned int*)(ws);                         // 2 MB
  float*          query   = (float*)(ws + (2u << 20));                   // 128 KB
  float*          scores  = (float*)(ws + (2u << 20) + (128u << 10));    // 256 KB

  float* ctx    = (float*)d_out;           // [32*1024]
  float* alphas = ctx + 32 * 1024;         // [2048*32]

  // allow 256 KB dynamic LDS for the WMMA kernel (ignore error if unneeded)
  (void)hipFuncSetAttribute((const void*)scores_kernel,
                            hipFuncAttributeMaxDynamicSharedMemorySize, 262144);

  cvt_wk_kernel<<<2048, 256, 0, stream>>>(Wk, wk_bf16);
  query_kernel<<<4, 256, 0, stream>>>(dec, Wq, query);
  scores_kernel<<<512, 128, 262144, stream>>>(
      enc, (const unsigned short*)wk_bf16, query, vv, scores);
  softmax_kernel<<<32, 256, 0, stream>>>(scores, mask, alphas);
  {
    dim3 grid(4, 32);
    context_kernel<<<grid, 256, 0, stream>>>(enc, alphas, ctx);
  }
}
